// LatticeMD_68212670595287
// MI455X (gfx1250) — matrix operand order repaired and checked by reading, not compiled, not run on hardware
//
#include <hip/hip_runtime.h>
#include <hip/hip_bf16.h>
#include <math.h>

// ---------------- problem constants ----------------
#define NBLK   4096          // 16*16*16 lattice blocks
#define HID    71            // ENC(64) + NM(7)
#define ENC    64
#define NMF    7
#define TSTEPS 8
#define LSTMIN 497           // 7 * 71
#define KCAT   576           // pad(497 + 71) to mult of 32
#define NCAT   320           // pad(4*71=284) to mult of 64
#define MROWS_ENC 32768      // T * NBLK

typedef __attribute__((ext_vector_type(16))) _Float16 v16h;
typedef __attribute__((ext_vector_type(8)))  float    v8f;

// ---- CDNA5 async global->LDS DMA (16 bytes / lane), tracked by ASYNCcnt ----
__device__ __forceinline__ void async_copy16(const void* g, void* l)
{
    // flat address of a __shared__ object carries the wave-relative LDS byte
    // offset in its low 32 bits (ISA 10.2), which is exactly the VDST operand.
    const unsigned loff = (unsigned)(unsigned long long)l;
    asm volatile("global_load_async_to_lds_b128 %0, %1, off"
                 :: "v"(loff), "v"(g)
                 : "memory");
}
__device__ __forceinline__ void wait_async0()
{
    asm volatile("s_wait_asynccnt 0x0" ::: "memory");
}

// ---------------- WMMA tiled GEMM: C = act(A @ B^T + bias) ----------------
// A: (M x K), row stride lda, dtype AT (float or _Float16)
// B: (Npad x K) row-major f16 ("weight" layout, already padded), row stride ldb
// C: (M x Ncols) within row stride ldc; f16 or f32 output
static constexpr int BM = 64, BN = 64, BK = 32;
static constexpr int LDSS = BK + 8;   // padded LDS stride (40 halfs = 80B = 5*16B)

template <typename AT, bool OUT_F16, bool TANH>
__global__ __launch_bounds__(256) void gemm_bias_act(
    const AT* __restrict__ A, int lda,
    const _Float16* __restrict__ Bmat, int ldb,
    const float* __restrict__ bias,
    void* __restrict__ C, int ldc,
    int Ncols, int K)
{
    __shared__ _Float16 As[BM * LDSS];
    __shared__ _Float16 Bs[BN * LDSS];

    const int tid  = threadIdx.x;
    const int lane = tid & 31;
    const int wave = tid >> 5;
    const int m0   = blockIdx.x * BM;
    const int n0   = blockIdx.y * BN;

    const int wm = (wave >> 1) * 16;   // 0,16,32,48
    const int wn = (wave & 1) * 32;    // 0,32

    v8f acc0 = {};
    v8f acc1 = {};

    const int r_ld  = tid >> 2;          // staging row this thread covers
    const int cb_ld = (tid & 3) * 8;     // staging col base (8 halfs = 16B)

    for (int k0 = 0; k0 < K; k0 += BK) {
        // ---- stage B tile: async DMA, f16 weights straight into LDS ----
        async_copy16(Bmat + (size_t)(n0 + r_ld) * ldb + k0 + cb_ld,
                     Bs + r_ld * LDSS + cb_ld);

        // ---- stage A tile ----
        if constexpr (__is_same(AT, _Float16)) {
            async_copy16(A + (size_t)(m0 + r_ld) * lda + k0 + cb_ld,
                         As + r_ld * LDSS + cb_ld);
        } else {
            // f32 source: load, convert, ds_store
            const AT* src = A + (size_t)(m0 + r_ld) * lda + k0 + cb_ld;
            _Float16* dst = As + r_ld * LDSS + cb_ld;
#pragma unroll
            for (int j = 0; j < 8; ++j) dst[j] = (_Float16)src[j];
        }
        wait_async0();
        __syncthreads();

        // ---- A fragment (16x32 f16, ISA 7.12.2 layout) ----
        v16h afrag;
        {
            const int row = wm + (lane & 15);
            const int kb  = (lane >> 4) * 8;
            const _Float16* ap = As + row * LDSS;
#pragma unroll
            for (int j = 0; j < 16; ++j) {
                const int kk = kb + j + ((j >= 8) ? 8 : 0);
                afrag[j] = ap[kk];
            }
        }
        // ---- B fragments (32x16 f16): lanes 0-15 K=0..7,16..23, 16-31 K=8..15,24..31 ----
        v16h bfrag0, bfrag1;
        {
            const int koff = (lane >> 4) * 8;
            const int nc   = wn + (lane & 15);
            const _Float16* bp0 = Bs + nc * LDSS + koff;
            const _Float16* bp1 = Bs + (nc + 16) * LDSS + koff;
#pragma unroll
            for (int j = 0; j < 16; ++j) { const int kk = j + ((j >= 8) ? 8 : 0); bfrag0[j] = bp0[kk]; bfrag1[j] = bp1[kk]; }
        }

        acc0 = __builtin_amdgcn_wmma_f32_16x16x32_f16(
            false, afrag, false, bfrag0, (short)0, acc0, false, false);
        acc1 = __builtin_amdgcn_wmma_f32_16x16x32_f16(
            false, afrag, false, bfrag1, (short)0, acc1, false, false);
        __syncthreads();
    }

    // ---- store: C/D layout VGPR v: M=v (lanes 0-15) / v+8 (lanes 16-31) ----
    const int nloc = lane & 15;
    const int mh   = (lane >> 4) * 8;
#pragma unroll
    for (int v = 0; v < 8; ++v) {
        const int mrow = m0 + wm + v + mh;
        {
            const int n = n0 + wn + nloc;
            if (n < Ncols) {
                float val = acc0[v] + bias[n];
                if (TANH) val = tanhf(val);
                if (OUT_F16) ((_Float16*)C)[(size_t)mrow * ldc + n] = (_Float16)val;
                else         ((float*)C)[(size_t)mrow * ldc + n]    = val;
            }
        }
        {
            const int n = n0 + wn + 16 + nloc;
            if (n < Ncols) {
                float val = acc1[v] + bias[n];
                if (TANH) val = tanhf(val);
                if (OUT_F16) ((_Float16*)C)[(size_t)mrow * ldc + n] = (_Float16)val;
                else         ((float*)C)[(size_t)mrow * ldc + n]    = val;
            }
        }
    }
}

// ---------------- helper kernels ----------------
__global__ void k_f32_to_f16(const float* __restrict__ s, _Float16* __restrict__ d, int n)
{
    int i = blockIdx.x * blockDim.x + threadIdx.x;
    if (i < n) d[i] = (_Float16)s[i];
}

// Wcat (NCAT x KCAT) f16: [W_ih | W_hh | 0], zero-padded rows beyond 284
__global__ void k_build_wcat(const float* __restrict__ Wih, const float* __restrict__ Whh,
                             _Float16* __restrict__ Wcat)
{
    int i = blockIdx.x * blockDim.x + threadIdx.x;
    if (i >= NCAT * KCAT) return;
    int n = i / KCAT, k = i % KCAT;
    float v = 0.f;
    if (n < 4 * HID) {
        if (k < LSTMIN)            v = Wih[n * LSTMIN + k];
        else if (k < LSTMIN + HID) v = Whh[n * HID + (k - LSTMIN)];
    }
    Wcat[i] = (_Float16)v;
}

__global__ void k_build_bcat(const float* __restrict__ bih, const float* __restrict__ bhh,
                             float* __restrict__ bcat)
{
    int i = blockIdx.x * blockDim.x + threadIdx.x;
    if (i >= NCAT) return;
    bcat[i] = (i < 4 * HID) ? (bih[i] + bhh[i]) : 0.f;
}

__global__ void k_copy_nm(const float* __restrict__ nm, float* __restrict__ enc)
{
    int i = blockIdx.x * blockDim.x + threadIdx.x;   // over T*NBLK*7
    if (i >= TSTEPS * NBLK * NMF) return;
    int j = i % NMF, b = i / NMF;
    enc[(size_t)b * HID + ENC + j] = nm[i];
}

__global__ void k_zero(float* __restrict__ p, int n)
{
    int i = blockIdx.x * blockDim.x + threadIdx.x;
    if (i < n) p[i] = 0.f;
}

// Build X (NBLK x KCAT) f16 for one timestep: 7 neighbor feature groups + h + pad
__global__ void k_gather_x(const float* __restrict__ enc_t,   // (NBLK, HID)
                           const float* __restrict__ h,       // (NBLK, HID)
                           _Float16* __restrict__ X)          // (NBLK, KCAT)
{
    const int b  = blockIdx.x;
    const int ix = b >> 8, iy = (b >> 4) & 15, iz = b & 15;
    int nb[7];
    nb[0] = b;
    nb[1] = (((ix + 15) & 15) << 8) | (iy << 4) | iz;  // roll +1 along x
    nb[2] = (((ix + 1)  & 15) << 8) | (iy << 4) | iz;  // roll -1 along x
    nb[3] = (ix << 8) | (((iy + 15) & 15) << 4) | iz;
    nb[4] = (ix << 8) | (((iy + 1)  & 15) << 4) | iz;
    nb[5] = (ix << 8) | (iy << 4) | ((iz + 15) & 15);
    nb[6] = (ix << 8) | (iy << 4) | ((iz + 1)  & 15);

    for (int col = threadIdx.x; col < KCAT; col += blockDim.x) {
        float v;
        if (col < LSTMIN) {
            const int j = col / HID, f = col % HID;
            v = enc_t[(size_t)nb[j] * HID + f];
        } else if (col < LSTMIN + HID) {
            v = h[(size_t)b * HID + (col - LSTMIN)];
        } else {
            v = 0.f;
        }
        X[(size_t)b * KCAT + col] = (_Float16)v;
    }
}

__device__ __forceinline__ float sigf(float x) { return 1.f / (1.f + expf(-x)); }

__global__ void k_lstm_cell(const float* __restrict__ gates,  // (NBLK, NCAT)
                            float* __restrict__ h, float* __restrict__ c)
{
    int i = blockIdx.x * blockDim.x + threadIdx.x;
    if (i >= NBLK * HID) return;
    const int b = i / HID, f = i % HID;
    const float* g = gates + (size_t)b * NCAT;
    const float gi = g[f], gf = g[HID + f], gg = g[2 * HID + f], go = g[3 * HID + f];
    const float cn = sigf(gf) * c[i] + sigf(gi) * tanhf(gg);
    c[i] = cn;
    h[i] = sigf(go) * tanhf(cn);
}

// tiny 7->16->16->7 MLP, one thread per lattice block
__global__ void k_nm_mlp(const float* __restrict__ h,
                         const float* __restrict__ W1, const float* __restrict__ b1,
                         const float* __restrict__ W2, const float* __restrict__ b2,
                         const float* __restrict__ W3, const float* __restrict__ b3,
                         float* __restrict__ out)
{
    int b = blockIdx.x * blockDim.x + threadIdx.x;
    if (b >= NBLK) return;
    float x0[7], x1[16], x2[16];
#pragma unroll
    for (int j = 0; j < 7; ++j) x0[j] = h[(size_t)b * HID + ENC + j];
#pragma unroll
    for (int o = 0; o < 16; ++o) {
        float s = b1[o];
#pragma unroll
        for (int j = 0; j < 7; ++j) s += W1[o * 7 + j] * x0[j];
        x1[o] = tanhf(s);
    }
#pragma unroll
    for (int o = 0; o < 16; ++o) {
        float s = b2[o];
#pragma unroll
        for (int j = 0; j < 16; ++j) s += W2[o * 16 + j] * x1[j];
        x2[o] = tanhf(s);
    }
#pragma unroll
    for (int o = 0; o < 7; ++o) {
        float s = b3[o];
#pragma unroll
        for (int j = 0; j < 16; ++j) s += W3[o * 16 + j] * x2[j];
        out[(size_t)b * 7 + o] = s;
    }
}

// ---------------- launcher ----------------
static inline size_t a256(size_t x) { return (x + 255) & ~(size_t)255; }

extern "C" void kernel_launch(void* const* d_in, const int* in_sizes, int n_in,
                              void* d_out, int out_size, void* d_ws, size_t ws_size,
                              hipStream_t stream)
{
    // inputs (pytree flattening of setup_inputs dict)
    const float* matter = (const float*)d_in[0];   // (8,4096,1024)
    const float* nonmat = (const float*)d_in[1];   // (8,4096,7)
    const float* eW1 = (const float*)d_in[3];   // 256x1024
    const float* eW2 = (const float*)d_in[4];   // 256x256
    const float* eW3 = (const float*)d_in[5];   // 64x256
    const float* eb1 = (const float*)d_in[6];
    const float* eb2 = (const float*)d_in[7];
    const float* eb3 = (const float*)d_in[8];
    const float* dW1 = (const float*)d_in[9];   // 256x64
    const float* dW2 = (const float*)d_in[10];  // 256x256
    const float* dW3 = (const float*)d_in[11];  // 1024x256
    const float* db1 = (const float*)d_in[12];
    const float* db2 = (const float*)d_in[13];
    const float* db3 = (const float*)d_in[14];
    const float* Wih = (const float*)d_in[15];  // 284x497
    const float* Whh = (const float*)d_in[16];  // 284x71
    const float* bih = (const float*)d_in[17];
    const float* bhh = (const float*)d_in[18];
    const float* nW1 = (const float*)d_in[19];  // 16x7
    const float* nW2 = (const float*)d_in[20];  // 16x16
    const float* nW3 = (const float*)d_in[21];  // 7x16
    const float* nb1 = (const float*)d_in[22];
    const float* nb2 = (const float*)d_in[23];
    const float* nb3 = (const float*)d_in[24];

    float* matter_out = (float*)d_out;                       // 4096*1024
    float* nonmat_out = (float*)d_out + (size_t)NBLK * 1024; // 4096*7

    // workspace layout
    char* ws = (char*)d_ws;
    size_t off = 0;
    auto take = [&](size_t bytes) { void* p = ws + off; off = a256(off + bytes); return p; };
    _Float16* wp_e1  = (_Float16*)take((size_t)256 * 1024 * 2);
    _Float16* wp_e2  = (_Float16*)take((size_t)256 * 256 * 2);
    _Float16* wp_e3  = (_Float16*)take((size_t)64 * 256 * 2);
    _Float16* wp_d1  = (_Float16*)take((size_t)256 * 64 * 2);
    _Float16* wp_d2  = (_Float16*)take((size_t)256 * 256 * 2);
    _Float16* wp_d3  = (_Float16*)take((size_t)1024 * 256 * 2);
    _Float16* wp_cat = (_Float16*)take((size_t)NCAT * KCAT * 2);
    float*    b_cat  = (float*)take((size_t)NCAT * 4);
    _Float16* eh1    = (_Float16*)take((size_t)MROWS_ENC * 256 * 2);
    _Float16* eh2    = (_Float16*)take((size_t)MROWS_ENC * 256 * 2);
    float*    encoded= (float*)take((size_t)TSTEPS * NBLK * HID * 4);
    _Float16* X      = (_Float16*)take((size_t)NBLK * KCAT * 2);
    float*    gates  = (float*)take((size_t)NBLK * NCAT * 4);
    float*    hbuf   = (float*)take((size_t)NBLK * HID * 4);
    float*    cbuf   = (float*)take((size_t)NBLK * HID * 4);
    _Float16* dh1    = (_Float16*)take((size_t)NBLK * 256 * 2);
    _Float16* dh2    = (_Float16*)take((size_t)NBLK * 256 * 2);
    (void)ws_size; (void)in_sizes; (void)n_in; (void)out_size;

    const int TB = 256;
    auto blks = [](int n, int t) { return (n + t - 1) / t; };

    // ---- pack weights to f16 ----
    k_f32_to_f16<<<blks(256 * 1024, TB), TB, 0, stream>>>(eW1, wp_e1, 256 * 1024);
    k_f32_to_f16<<<blks(256 * 256, TB),  TB, 0, stream>>>(eW2, wp_e2, 256 * 256);
    k_f32_to_f16<<<blks(64 * 256, TB),   TB, 0, stream>>>(eW3, wp_e3, 64 * 256);
    k_f32_to_f16<<<blks(256 * 64, TB),   TB, 0, stream>>>(dW1, wp_d1, 256 * 64);
    k_f32_to_f16<<<blks(256 * 256, TB),  TB, 0, stream>>>(dW2, wp_d2, 256 * 256);
    k_f32_to_f16<<<blks(1024 * 256, TB), TB, 0, stream>>>(dW3, wp_d3, 1024 * 256);
    k_build_wcat<<<blks(NCAT * KCAT, TB), TB, 0, stream>>>(Wih, Whh, wp_cat);
    k_build_bcat<<<blks(NCAT, TB), TB, 0, stream>>>(bih, bhh, b_cat);

    // ---- encoder MLP (32768 rows) ----
    gemm_bias_act<float, true, true><<<dim3(MROWS_ENC / 64, 256 / 64), 256, 0, stream>>>(
        matter, 1024, wp_e1, 1024, eb1, eh1, 256, 256, 1024);
    gemm_bias_act<_Float16, true, true><<<dim3(MROWS_ENC / 64, 256 / 64), 256, 0, stream>>>(
        eh1, 256, wp_e2, 256, eb2, eh2, 256, 256, 256);
    gemm_bias_act<_Float16, false, false><<<dim3(MROWS_ENC / 64, 1), 256, 0, stream>>>(
        eh2, 256, wp_e3, 256, eb3, encoded, HID, 64, 256);   // into cols 0..63, stride 71
    k_copy_nm<<<blks(TSTEPS * NBLK * NMF, TB), TB, 0, stream>>>(nonmat, encoded);

    // ---- LSTM over 8 steps: gates = [x, h] @ Wcat^T + bcat ----
    k_zero<<<blks(NBLK * HID, TB), TB, 0, stream>>>(hbuf, NBLK * HID);
    k_zero<<<blks(NBLK * HID, TB), TB, 0, stream>>>(cbuf, NBLK * HID);
    for (int t = 0; t < TSTEPS; ++t) {
        k_gather_x<<<NBLK, 128, 0, stream>>>(encoded + (size_t)t * NBLK * HID, hbuf, X);
        gemm_bias_act<_Float16, false, false><<<dim3(NBLK / 64, NCAT / 64), 256, 0, stream>>>(
            X, KCAT, wp_cat, KCAT, b_cat, gates, NCAT, NCAT, KCAT);
        k_lstm_cell<<<blks(NBLK * HID, TB), TB, 0, stream>>>(gates, hbuf, cbuf);
    }

    // ---- decoder MLP on h[:, :64] ----
    gemm_bias_act<float, true, true><<<dim3(NBLK / 64, 256 / 64), 256, 0, stream>>>(
        hbuf, HID, wp_d1, 64, db1, dh1, 256, 256, 64);
    gemm_bias_act<_Float16, true, true><<<dim3(NBLK / 64, 256 / 64), 256, 0, stream>>>(
        dh1, 256, wp_d2, 256, db2, dh2, 256, 256, 256);
    gemm_bias_act<_Float16, false, false><<<dim3(NBLK / 64, 1024 / 64), 256, 0, stream>>>(
        dh2, 256, wp_d3, 256, db3, matter_out, 1024, 1024, 256);

    // ---- nonmatter head on h[:, 64:71] ----
    k_nm_mlp<<<blks(NBLK, TB), TB, 0, stream>>>(hbuf, nW1, nb1, nW2, nb2, nW3, nb3, nonmat_out);
}
